// MainLayer_63556926046507
// MI455X (gfx1250) — matrix operand order repaired and checked by reading, not compiled, not run on hardware
//
#include <hip/hip_runtime.h>
#include <math.h>
#include <stdint.h>

// ---------------------------------------------------------------------------
// Problem constants (from the reference)
// ---------------------------------------------------------------------------
#define D_     1024
#define HD_    64
#define NH_    16
#define H_     16
#define ET_    16
#define N_     128
#define E_     4096
#define RE_    4
#define S_     2048
#define T_     2048
#define BC_    16
#define EPS_   1e-5f
#define LOG2_THETA_ 13.287712379549449f   // log2(10000)

typedef __attribute__((ext_vector_type(16))) _Float16 v16h;
typedef __attribute__((ext_vector_type(8)))  float    v8f;

#define WMMA_F16(a, b, c) \
  __builtin_amdgcn_wmma_f32_16x16x32_f16(false, (a), false, (b), (short)0, (c), false, false)

#if defined(__AMDGCN__) && __has_builtin(__builtin_amdgcn_tensor_load_to_lds)
#define USE_TDM 1
#else
#define USE_TDM 0
#endif

static __device__ __forceinline__ int laneid() { return threadIdx.x & 31; }

// A-fragment (16x32 f16, M x K). Row-major source: lane holds row m = lane&15;
// VGPR0-3: K = half*16 + {0..7}, VGPR4-7: K = half*16 + 8 + {0..7} (pairs).
static __device__ __forceinline__ v16h load_a_frag(const _Float16* base, int ld) {
  const int l = laneid();
  const int m = l & 15, half = l >> 4;
  union { v16h v; uint32_t u[8]; } f;
  const uint32_t* p0 = (const uint32_t*)(base + (size_t)m * ld + half * 16);
  const uint32_t* p1 = (const uint32_t*)(base + (size_t)m * ld + half * 16 + 8);
  f.u[0] = p0[0]; f.u[1] = p0[1]; f.u[2] = p0[2]; f.u[3] = p0[3];
  f.u[4] = p1[0]; f.u[5] = p1[1]; f.u[6] = p1[2]; f.u[7] = p1[3];
  return f.v;
}

// B-fragment (32x16 f16, K x N). Source has each column's K contiguous
// (ld between columns): lane holds column n = lane&15, K = half*16 + {0..15}.
static __device__ __forceinline__ v16h load_b_frag(const _Float16* base, int ld) {
  const int l = laneid();
  const int n = l & 15, half = l >> 4;
  union { v16h v; uint32_t u[8]; } f;
  const uint32_t* p = (const uint32_t*)(base + (size_t)n * ld + half * 16);
#pragma unroll
  for (int v = 0; v < 8; ++v) f.u[v] = p[v];
  return f.v;
}

// ---------------------------------------------------------------------------
// Tensor Data Mover: 2-D tile (tile_dim0 = 32 f16 along K, tile_dim1 = 128
// rows, row stride = ld elements) from global into LDS. One wave issues; the
// whole workgroup synchronizes on TENSORcnt + barrier.
// ---------------------------------------------------------------------------
#if USE_TDM
typedef __attribute__((ext_vector_type(4))) uint32_t u32x4;
typedef __attribute__((ext_vector_type(8))) int      i32x8;
typedef __attribute__((ext_vector_type(4))) int      i32x4;

static __device__ __forceinline__ void tdm_load_tile(uint32_t lds_addr,
                                                     const _Float16* gptr,
                                                     int row_stride_elems) {
  const uint64_t ga = (uint64_t)(uintptr_t)gptr;
  u32x4 g0;
  g0[0] = 1u;                                            // count=1, user desc
  g0[1] = lds_addr;                                      // LDS byte address
  g0[2] = (uint32_t)ga;                                  // global_addr[31:0]
  g0[3] = (uint32_t)((ga >> 32) & 0x01FFFFFFu) | (2u << 30); // addr[56:32], type=2
  i32x8 g1;
  g1[0] = 0x00010000;                 // workgroup_mask=0, data_size=1 (2 bytes)
  g1[1] = 0;                          // abar addr=0, tensor_dim0[15:0]=0
  g1[2] = 0x00000010;                 // tensor_dim0 hi=16 (=1M), tensor_dim1 lo=0
  g1[3] = 0x00000010 | (32 << 16);    // tensor_dim1 hi=16 (=1M), tile_dim0=32
  g1[4] = 128;                        // tile_dim1=128, tile_dim2=0
  g1[5] = row_stride_elems;           // tensor_dim0_stride low32 (elements)
  g1[6] = 0;
  g1[7] = 0;
  i32x4 z = {0, 0, 0, 0};
#if defined(__clang_major__) && __clang_major__ >= 23
  i32x8 z8 = {0, 0, 0, 0, 0, 0, 0, 0};
  __builtin_amdgcn_tensor_load_to_lds(g0, g1, z, z, z8, 0);
#else
  __builtin_amdgcn_tensor_load_to_lds(g0, g1, z, z, 0);
#endif
}
#endif

// ---------------------------------------------------------------------------
// K1: f32 -> f16 transpose (weights become [N][K], K contiguous for B frags)
// ---------------------------------------------------------------------------
__global__ void convert_transpose_kernel(const float* __restrict__ in,
                                         _Float16* __restrict__ out,
                                         int R, int C) {
  size_t i = (size_t)blockIdx.x * 256 + threadIdx.x;
  size_t total = (size_t)R * C;
  if (i >= total) return;
  int r = (int)(i / C), c = (int)(i % C);
  out[(size_t)c * R + r] = (_Float16)in[i];
}

// ---------------------------------------------------------------------------
// K2: RMSNorm (fp32 stats) -> f16 activations
// ---------------------------------------------------------------------------
__global__ void rmsnorm_kernel(const float* __restrict__ x,
                               const float* __restrict__ w,
                               _Float16* __restrict__ out) {
  __shared__ float red[8];
  __shared__ float s_scale;
  const int t = blockIdx.x;
  const int tid = threadIdx.x;
  const float* xr = x + (size_t)t * D_;
  float s = 0.f;
  for (int i = tid; i < D_; i += 256) { float v = xr[i]; s += v * v; }
#pragma unroll
  for (int off = 16; off; off >>= 1) s += __shfl_xor(s, off, 32);
  if ((tid & 31) == 0) red[tid >> 5] = s;
  __syncthreads();
  if (tid == 0) {
    float tot = 0.f;
#pragma unroll
    for (int i = 0; i < 8; ++i) tot += red[i];
    s_scale = rsqrtf(tot / (float)D_ + EPS_);
  }
  __syncthreads();
  const float sc = s_scale;
  for (int i = tid; i < D_; i += 256)
    out[(size_t)t * D_ + i] = (_Float16)(xr[i] * sc * w[i]);
}

// ---------------------------------------------------------------------------
// K3: WMMA GEMM with TDM-staged, double-buffered LDS tiles.
// C[M,N] = A[M,K] * Bt[N,K]^T (+ optional f32 residual).
// block = 256 threads (8 waves); block tile 128x128; K-step 32.
// Wave w computes rows [w*16, w*16+16) x all 128 cols (8 accumulators).
// ---------------------------------------------------------------------------
__global__ void gemm_f16_kernel(const _Float16* __restrict__ A,
                                const _Float16* __restrict__ Bt,
                                float* __restrict__ C,
                                const float* __restrict__ res,
                                int M, int N, int K) {
  __shared__ _Float16 lds_a[2][128 * 32];   // [buf][row][k]
  __shared__ _Float16 lds_b[2][128 * 32];   // [buf][col][k]
  const int tid = threadIdx.x;
  const int wave = tid >> 5;
  const int m0 = blockIdx.y * 128;
  const int n0 = blockIdx.x * 128;

  auto stage = [&](int buf, int kt) {
#if USE_TDM
    if (tid < 32) {   // wave 0 issues both DMA descriptors
      tdm_load_tile((uint32_t)(uintptr_t)&lds_a[buf][0], A + (size_t)m0 * K + kt, K);
      tdm_load_tile((uint32_t)(uintptr_t)&lds_b[buf][0], Bt + (size_t)n0 * K + kt, K);
    }
#else
    for (int i = tid; i < 128 * 16; i += 256) {   // copy as dword pairs
      const int r = i >> 4, kk = (i & 15) << 1;
      *(uint32_t*)&lds_a[buf][r * 32 + kk] =
          *(const uint32_t*)&A[(size_t)(m0 + r) * K + kt + kk];
      *(uint32_t*)&lds_b[buf][r * 32 + kk] =
          *(const uint32_t*)&Bt[(size_t)(n0 + r) * K + kt + kk];
    }
#endif
  };

  stage(0, 0);
#if USE_TDM
  if (tid < 32) __builtin_amdgcn_s_wait_tensorcnt(0);
#endif
  __syncthreads();

  v8f acc[8] = {};
  for (int kt = 0; kt < K; kt += 32) {
    const int cur = (kt >> 5) & 1;
    if (kt + 32 < K) stage(cur ^ 1, kt + 32);   // overlap DMA with WMMA
    v16h a = load_a_frag(&lds_a[cur][wave * 16 * 32], 32);
#pragma unroll
    for (int nt = 0; nt < 8; ++nt) {
      v16h b = load_b_frag(&lds_b[cur][nt * 16 * 32], 32);
      acc[nt] = WMMA_F16(a, b, acc[nt]);
    }
#if USE_TDM
    if (tid < 32) __builtin_amdgcn_s_wait_tensorcnt(0);
#endif
    __syncthreads();
  }

  const int l = laneid();
  const int n = l & 15, mb = (l >> 4) * 8;
#pragma unroll
  for (int nt = 0; nt < 8; ++nt)
#pragma unroll
    for (int r = 0; r < 8; ++r) {
      size_t row = (size_t)(m0 + wave * 16 + mb + r);
      size_t col = (size_t)(n0 + nt * 16 + n);
      float v = acc[nt][r];
      if (res) v += res[row * N + col];
      C[row * N + col] = v;
    }
}

// ---------------------------------------------------------------------------
// K4: l2norm + RoPE on Q,K; split V (transposed [h][d][S]) — all f16 out
// grid S, block 512 = 16 waves, wave = head, lane = dim-pair index
// ---------------------------------------------------------------------------
__global__ void l2rope_kernel(const float* __restrict__ qkv,
                              _Float16* __restrict__ Qh,
                              _Float16* __restrict__ Kh,
                              _Float16* __restrict__ Vt) {
  const int s = blockIdx.x;
  const int h = threadIdx.x >> 5;
  const int d = threadIdx.x & 31;
  const float* base = qkv + (size_t)s * (3 * D_) + h * HD_;
  float q1 = base[d], q2 = base[d + 32];
  float k1 = base[D_ + d], k2 = base[D_ + d + 32];
  float v1 = base[2 * D_ + d], v2 = base[2 * D_ + d + 32];
  float sq = q1 * q1 + q2 * q2;
  float sk = k1 * k1 + k2 * k2;
#pragma unroll
  for (int off = 16; off; off >>= 1) {
    sq += __shfl_xor(sq, off, 32);
    sk += __shfl_xor(sk, off, 32);
  }
  const float qs = 1.f / fmaxf(sqrtf(sq), EPS_);
  const float ks = 1.f / fmaxf(sqrtf(sk), EPS_);
  q1 *= qs; q2 *= qs; k1 *= ks; k2 *= ks;
  const float fr = (float)s * exp2f(-(float)d * (LOG2_THETA_ / 32.f));
  const float c = __cosf(fr), sn = __sinf(fr);
  const size_t qo = ((size_t)h * S_ + s) * HD_;
  Qh[qo + d]      = (_Float16)(q1 * c + q2 * sn);
  Qh[qo + d + 32] = (_Float16)(-q1 * sn + q2 * c);
  Kh[qo + d]      = (_Float16)(k1 * c + k2 * sn);
  Kh[qo + d + 32] = (_Float16)(-k1 * sn + k2 * c);
  Vt[(size_t)h * HD_ * S_ + (size_t)d * S_ + s]        = (_Float16)v1;
  Vt[(size_t)h * HD_ * S_ + (size_t)(d + 32) * S_ + s] = (_Float16)v2;
}

// ---------------------------------------------------------------------------
// K5: causal flash attention, WMMA f16. grid (S/64, NH), block 128 (4 waves).
// Each wave owns 16 query rows; key blocks of 32; P transposed via LDS.
// ---------------------------------------------------------------------------
__global__ void flash_attn_kernel(const _Float16* __restrict__ Q,
                                  const _Float16* __restrict__ K,
                                  const _Float16* __restrict__ Vt,
                                  _Float16* __restrict__ Oh) {
  __shared__ _Float16 pbuf[4][16 * 32];
  const int h = blockIdx.y;
  const int wave = threadIdx.x >> 5;
  const int l = laneid();
  const int n = l & 15, mb = (l >> 4) * 8;
  const int q0 = blockIdx.x * 64 + wave * 16;
  const float NEGINF = -__builtin_inff();

  const _Float16* Qb = Q + ((size_t)h * S_ + q0) * HD_;
  v16h qa0 = load_a_frag(Qb, HD_);       // d = 0..31
  v16h qa1 = load_a_frag(Qb + 32, HD_);  // d = 32..63

  v8f o[4] = {};
  float m[8], lsum[8];
#pragma unroll
  for (int r = 0; r < 8; ++r) { m[r] = NEGINF; lsum[r] = 0.f; }

  const int nkb = (q0 + 16 + 31) / 32;   // causal: keys <= q0+15
  for (int kb = 0; kb < nkb; ++kb) {
    const int kbase = kb * 32;
    const _Float16* Kb = K + ((size_t)h * S_ + kbase) * HD_;
    v8f s0 = {}, s1 = {};
    s0 = WMMA_F16(qa0, load_b_frag(Kb, HD_), s0);
    s0 = WMMA_F16(qa1, load_b_frag(Kb + 32, HD_), s0);
    s1 = WMMA_F16(qa0, load_b_frag(Kb + (size_t)16 * HD_, HD_), s1);
    s1 = WMMA_F16(qa1, load_b_frag(Kb + (size_t)16 * HD_ + 32, HD_), s1);

    float p0v[8], p1v[8];
#pragma unroll
    for (int r = 0; r < 8; ++r) {
      const int qrow = q0 + mb + r;
      float v0 = s0[r] * 0.125f; if (kbase + n > qrow)      v0 = NEGINF;
      float v1 = s1[r] * 0.125f; if (kbase + 16 + n > qrow) v1 = NEGINF;
      float rmax = fmaxf(v0, v1);
      rmax = fmaxf(rmax, __shfl_xor(rmax, 1, 32));
      rmax = fmaxf(rmax, __shfl_xor(rmax, 2, 32));
      rmax = fmaxf(rmax, __shfl_xor(rmax, 4, 32));
      rmax = fmaxf(rmax, __shfl_xor(rmax, 8, 32));
      const float mn = fmaxf(m[r], rmax);
      const float fac = __expf(m[r] - mn);   // exp(-inf)=0 on first block
      const float p0 = __expf(v0 - mn);
      const float p1 = __expf(v1 - mn);
      float rs = p0 + p1;
      rs += __shfl_xor(rs, 1, 32);
      rs += __shfl_xor(rs, 2, 32);
      rs += __shfl_xor(rs, 4, 32);
      rs += __shfl_xor(rs, 8, 32);
      lsum[r] = lsum[r] * fac + rs;
      m[r] = mn;
#pragma unroll
      for (int j = 0; j < 4; ++j) o[j][r] *= fac;
      p0v[r] = p0; p1v[r] = p1;
    }
    // stage P (C-layout) -> LDS, re-read as A-fragment (16x32)
#pragma unroll
    for (int r = 0; r < 8; ++r) {
      pbuf[wave][(mb + r) * 32 + n]      = (_Float16)p0v[r];
      pbuf[wave][(mb + r) * 32 + 16 + n] = (_Float16)p1v[r];
    }
    asm volatile("s_wait_dscnt 0x0" ::: "memory");   // same-wave LDS RAW
    v16h pa = load_a_frag(&pbuf[wave][0], 32);
    const _Float16* Vb = Vt + (size_t)h * HD_ * S_ + kbase;
#pragma unroll
    for (int j = 0; j < 4; ++j)
      o[j] = WMMA_F16(pa, load_b_frag(Vb + (size_t)(j * 16) * S_, S_), o[j]);
  }
#pragma unroll
  for (int j = 0; j < 4; ++j)
#pragma unroll
    for (int r = 0; r < 8; ++r) {
      const size_t row = (size_t)(q0 + mb + r);
      Oh[row * D_ + h * HD_ + j * 16 + n] = (_Float16)(o[j][r] / lsum[r]);
    }
}

// ---------------------------------------------------------------------------
// K6: router at *gathered* experts only: per token-block GEMM 128x256x1024
// (N-tile nt == expert slot, 16 cols == heads) + sigmoid/head-prob epilogue.
// grid BC_, block 256 (8 waves, wave = 16-token row group).
// ---------------------------------------------------------------------------
__global__ void routing_kernel(const _Float16* __restrict__ xf,
                               const float* __restrict__ keys,        // [KH][D][E]
                               const int* __restrict__ idxs,          // [BC][ET]
                               const float* __restrict__ scores,      // [T][ET][H]
                               const float* __restrict__ head_probs,  // [RE][E][H]
                               const float* __restrict__ score_probs, // [2][RE][E][H]
                               float* __restrict__ wgt) {             // [T][ET]
  __shared__ _Float16 bt[256 * 32];   // 16 KB: [n][k] tile, K pairs contiguous
  __shared__ int eidx[ET_];
  const int b = blockIdx.x;
  const int tid = threadIdx.x;
  if (tid < ET_) eidx[tid] = idxs[b * ET_ + tid];
  __syncthreads();
  const int wave = tid >> 5;
  const int t0 = b * N_ + wave * 16;
  const int e_l = tid >> 4, h_l = tid & 15;
  const float* kcol = keys + (size_t)(h_l >> 1) * D_ * E_ + eidx[e_l];

  v8f acc[16] = {};
  for (int kt = 0; kt < D_; kt += 32) {
    __syncthreads();
#pragma unroll
    for (int kk = 0; kk < 32; ++kk)
      bt[tid * 32 + kk] = (_Float16)kcol[(size_t)(kt + kk) * E_];
    __syncthreads();
    v16h a = load_a_frag(xf + (size_t)t0 * D_ + kt, D_);
#pragma unroll
    for (int nt = 0; nt < 16; ++nt)
      acc[nt] = WMMA_F16(a, load_b_frag(bt + nt * 16 * 32, 32), acc[nt]);
  }
  const int l = laneid();
  const int hl = l & 15, mb = (l >> 4) * 8;
#pragma unroll
  for (int nt = 0; nt < 16; ++nt) {
    const int expert = eidx[nt];
    const int re = nt >> 2;
    const size_t off = ((size_t)re * E_ + expert) * H_ + hl;
    const float sp0 = score_probs[off];
    const float sp1 = score_probs[(size_t)RE_ * E_ * H_ + off];
    const float hp  = head_probs[off];
#pragma unroll
    for (int r = 0; r < 8; ++r) {
      const int t = t0 + mb + r;
      const float sc = scores[((size_t)t * ET_ + nt) * H_ + hl];
      const float z = sp0 * acc[nt][r] + sp1 * sc;
      float contrib = hp / (1.f + __expf(-z));
      contrib += __shfl_xor(contrib, 1, 32);
      contrib += __shfl_xor(contrib, 2, 32);
      contrib += __shfl_xor(contrib, 4, 32);
      contrib += __shfl_xor(contrib, 8, 32);
      if (hl == 0) wgt[(size_t)t * ET_ + nt] = contrib;
    }
  }
}

// ---------------------------------------------------------------------------
// K7: expert activations  act[t][e] = silu(h0)*h1*w  (tiny FLOPs: VALU dots)
// grid T, block 32 (1 wave per token)
// ---------------------------------------------------------------------------
__global__ void expert_act_kernel(const _Float16* __restrict__ xf,
                                  const float* __restrict__ experts, // [3][E][D]
                                  const int* __restrict__ idxs,
                                  const float* __restrict__ wgt,
                                  float* __restrict__ act) {
  const int t = blockIdx.x;
  const int lane = threadIdx.x;
  const int b = t >> 7;  // N_=128
  const _Float16* xr = xf + (size_t)t * D_;
  float xv[32];
#pragma unroll
  for (int i = 0; i < 32; ++i) xv[i] = (float)xr[lane + i * 32];
  for (int e = 0; e < ET_; ++e) {
    const int expert = idxs[b * ET_ + e];
    const float* w0 = experts + (size_t)expert * D_;
    const float* w1 = experts + (size_t)E_ * D_ + (size_t)expert * D_;
    float a0 = 0.f, a1 = 0.f;
#pragma unroll
    for (int i = 0; i < 32; ++i) {
      const int d = lane + i * 32;
      a0 += xv[i] * w0[d];
      a1 += xv[i] * w1[d];
    }
#pragma unroll
    for (int off = 16; off; off >>= 1) {
      a0 += __shfl_xor(a0, off, 32);
      a1 += __shfl_xor(a1, off, 32);
    }
    if (lane == 0) {
      const float sig = 1.f / (1.f + __expf(-a0));
      act[(size_t)t * ET_ + e] = (a0 * sig) * a1 * wgt[(size_t)t * ET_ + e];
    }
  }
}

// ---------------------------------------------------------------------------
// K8: out[t][d] = x_ffn_input[t][d] + sum_e act[t][e] * W2[idx][d]
// grid T, block 256
// ---------------------------------------------------------------------------
__global__ void final_out_kernel(const float* __restrict__ xffn,
                                 const float* __restrict__ experts,
                                 const int* __restrict__ idxs,
                                 const float* __restrict__ act,
                                 float* __restrict__ out) {
  __shared__ float a_s[ET_];
  __shared__ int e_s[ET_];
  const int t = blockIdx.x;
  const int tid = threadIdx.x;
  const int b = t >> 7;
  if (tid < ET_) {
    a_s[tid] = act[(size_t)t * ET_ + tid];
    e_s[tid] = idxs[b * ET_ + tid];
  }
  __syncthreads();
  const float* W2 = experts + (size_t)2 * E_ * D_;
  for (int d = tid; d < D_; d += 256) {
    float accv = xffn[(size_t)t * D_ + d];
#pragma unroll
    for (int e = 0; e < ET_; ++e)
      accv += a_s[e] * W2[(size_t)e_s[e] * D_ + d];
    out[(size_t)t * D_ + d] = accv;
  }
}

// ---------------------------------------------------------------------------
// Host launcher
// ---------------------------------------------------------------------------
extern "C" void kernel_launch(void* const* d_in, const int* in_sizes, int n_in,
                              void* d_out, int out_size, void* d_ws, size_t ws_size,
                              hipStream_t stream) {
  (void)in_sizes; (void)n_in; (void)out_size; (void)ws_size;
  const float* x_input     = (const float*)d_in[0];
  const int*   indices     = (const int*)d_in[1];
  const float* scores      = (const float*)d_in[2];
  const float* attn_w      = (const float*)d_in[3];
  const float* attn_out_w  = (const float*)d_in[4];
  const float* attn_norm_w = (const float*)d_in[5];
  const float* ffn_norm_w  = (const float*)d_in[6];
  const float* ffn_experts = (const float*)d_in[7];
  const float* keys        = (const float*)d_in[8];
  const float* head_probs  = (const float*)d_in[9];
  const float* score_probs = (const float*)d_in[10];
  float* out = (float*)d_out;

  char* ws = (char*)d_ws;
  size_t off = 0;
  auto carve = [&](size_t bytes) { char* p = ws + off; off += (bytes + 255) & ~(size_t)255; return p; };
  _Float16* xn_h    = (_Float16*)carve((size_t)T_ * D_ * 2);       // rmsnorm(x) f16
  _Float16* wqkv_t  = (_Float16*)carve((size_t)3 * D_ * D_ * 2);   // attn_w^T f16 [3D][D]
  _Float16* wout_t  = (_Float16*)carve((size_t)D_ * D_ * 2);       // attn_out_w^T f16
  float*    qkv     = (float*)   carve((size_t)T_ * 3 * D_ * 4);   // QKV f32
  _Float16* q_h     = (_Float16*)carve((size_t)T_ * D_ * 2);       // [h][s][64]
  _Float16* k_h     = (_Float16*)carve((size_t)T_ * D_ * 2);       // [h][s][64]
  _Float16* v_h     = (_Float16*)carve((size_t)T_ * D_ * 2);       // [h][64][s]
  _Float16* xattn_h = (_Float16*)carve((size_t)T_ * D_ * 2);       // attn out f16
  float*    xffn    = (float*)   carve((size_t)T_ * D_ * 4);       // x_ffn_input f32
  _Float16* xf_h    = (_Float16*)carve((size_t)T_ * D_ * 2);       // rmsnorm2 f16
  float*    wgt     = (float*)   carve((size_t)T_ * ET_ * 4);      // router weights
  float*    act     = (float*)   carve((size_t)T_ * ET_ * 4);      // expert act

  convert_transpose_kernel<<<(D_ * 3 * D_ + 255) / 256, 256, 0, stream>>>(attn_w, wqkv_t, D_, 3 * D_);
  convert_transpose_kernel<<<(D_ * D_ + 255) / 256, 256, 0, stream>>>(attn_out_w, wout_t, D_, D_);
  rmsnorm_kernel<<<T_, 256, 0, stream>>>(x_input, attn_norm_w, xn_h);
  // QKV GEMM (2048 x 3072 x 1024), 128x128 block tiles
  gemm_f16_kernel<<<dim3(3 * D_ / 128, T_ / 128), 256, 0, stream>>>(xn_h, wqkv_t, qkv, nullptr, T_, 3 * D_, D_);
  l2rope_kernel<<<S_, NH_ * 32, 0, stream>>>(qkv, q_h, k_h, v_h);
  flash_attn_kernel<<<dim3(S_ / 64, NH_), 128, 0, stream>>>(q_h, k_h, v_h, xattn_h);
  // out-proj GEMM + residual(x_input) -> x_ffn_input
  gemm_f16_kernel<<<dim3(D_ / 128, T_ / 128), 256, 0, stream>>>(xattn_h, wout_t, xffn, x_input, T_, D_, D_);
  rmsnorm_kernel<<<T_, 256, 0, stream>>>(xffn, ffn_norm_w, xf_h);
  routing_kernel<<<BC_, 256, 0, stream>>>(xf_h, keys, indices, scores, head_probs, score_probs, wgt);
  expert_act_kernel<<<T_, 32, 0, stream>>>(xf_h, ffn_experts, indices, wgt, act);
  final_out_kernel<<<T_, 256, 0, stream>>>(xffn, ffn_experts, indices, act, out);
}
